// Attention_31044023615903
// MI455X (gfx1250) — matrix operand order repaired and checked by reading, not compiled, not run on hardware
//
#include <hip/hip_runtime.h>
#include <hip/hip_bf16.h>

#define D_MODEL 1024
#define N_HEADS 16
#define HEAD_DIM 64
#define BATCH 2
#define SEQ 2048
#define NTOK (BATCH * SEQ)      // 4096 tokens
#define QKV_N (3 * D_MODEL)     // 3072

typedef __bf16 bf16_t;
typedef __attribute__((ext_vector_type(16))) bf16_t v16bf;
typedef __attribute__((ext_vector_type(8)))  bf16_t v8bf;
typedef __attribute__((ext_vector_type(8)))  float  v8f;

union Frag16 { v16bf v; v8bf h[2]; };

// A-fragment (16x32 bf16, MxK), row-major source, per-lane row m = lane&15.
// e=0..15 -> K = k0 + half*16 + e (contiguous 32B)
__device__ inline v16bf load_a_frag(const bf16_t* row, int k0, int half) {
  Frag16 f;
  f.h[0] = *(const v8bf*)(row + k0 + half * 16);
  f.h[1] = *(const v8bf*)(row + k0 + half * 16 + 8);
  return f.v;
}

// B-fragment (32x16 bf16, KxN) sourced from Bt (N-major, rows of length ldb):
// lane column n = lane&15, K = k0 + half*16 + e (contiguous 32B)
__device__ inline v16bf load_b_frag(const bf16_t* row, int k0, int half) {
  Frag16 f;
  const bf16_t* p = row + k0 + half * 16;
  f.h[0] = *(const v8bf*)(p);
  f.h[1] = *(const v8bf*)(p + 8);
  return f.v;
}

__device__ inline v8f wmma_bf16(v16bf a, v16bf b, v8f c) {
  return __builtin_amdgcn_wmma_f32_16x16x32_bf16(false, a, false, b,
                                                 (short)0, c, false, false);
}

// ---------------- conversion kernels ----------------
__global__ void k_f32_to_bf16(const float* __restrict__ in,
                              bf16_t* __restrict__ out, int n) {
  int i = blockIdx.x * 256 + threadIdx.x;
  if (i < n) out[i] = (bf16_t)in[i];
}

// in: [rows, cols] f32 row-major -> out: [cols, rows] bf16 row-major
__global__ void k_transpose_to_bf16(const float* __restrict__ in,
                                    bf16_t* __restrict__ out,
                                    int rows, int cols) {
  int i = blockIdx.x * 256 + threadIdx.x;
  if (i >= rows * cols) return;
  int r = i / cols, c = i - r * cols;
  out[(size_t)c * rows + r] = (bf16_t)in[i];
}

// vt[((b*H+h)*64 + d)*SEQ + j] = V[b, token j, head h, d] from qkv buffer
__global__ void k_build_vt(const bf16_t* __restrict__ qkvb,
                           bf16_t* __restrict__ vt) {
  int i = blockIdx.x * 256 + threadIdx.x;   // B*H*64*SEQ = 4M elements
  int j  = i & (SEQ - 1);
  int t  = i >> 11;        // (b*H+h)*64 + d
  int d  = t & 63;
  int bh = t >> 6;
  int b  = bh >> 4;
  int h  = bh & 15;
  vt[i] = qkvb[((size_t)(b * SEQ + j)) * QKV_N + 2 * D_MODEL + h * HEAD_DIM + d];
}

// ---------------- bf16 WMMA GEMM ----------------
// C[M,N] = A[M,K] * Bt[N,K]^T + bias ; block = 8 waves, wave tile 64x64,
// block tile 512x64.  Writes bf16 (outb) or f32 (outf).
__global__ void __launch_bounds__(256)
k_gemm_bf16(const bf16_t* __restrict__ A, int lda,
            const bf16_t* __restrict__ Bt, int ldb,
            const float* __restrict__ bias,
            bf16_t* __restrict__ outb, float* __restrict__ outf,
            int ldo, int K) {
  const int lane = threadIdx.x & 31;
  const int wave = threadIdx.x >> 5;
  const int half = lane >> 4;
  const int ln   = lane & 15;
  const int row0 = blockIdx.y * 512 + wave * 64;
  const int col0 = blockIdx.x * 64;

  v8f acc[4][4];
#pragma unroll
  for (int mt = 0; mt < 4; ++mt)
#pragma unroll
    for (int nt = 0; nt < 4; ++nt) acc[mt][nt] = (v8f){};

  const bf16_t* arow[4];
#pragma unroll
  for (int mt = 0; mt < 4; ++mt)
    arow[mt] = A + (size_t)(row0 + mt * 16 + ln) * lda;

  for (int k0 = 0; k0 < K; k0 += 32) {
    v16bf a0 = load_a_frag(arow[0], k0, half);
    v16bf a1 = load_a_frag(arow[1], k0, half);
    v16bf a2 = load_a_frag(arow[2], k0, half);
    v16bf a3 = load_a_frag(arow[3], k0, half);
#pragma unroll
    for (int nt = 0; nt < 4; ++nt) {
      const bf16_t* brow = Bt + (size_t)(col0 + nt * 16 + ln) * ldb;
      v16bf b = load_b_frag(brow, k0, half);
      acc[0][nt] = wmma_bf16(a0, b, acc[0][nt]);
      acc[1][nt] = wmma_bf16(a1, b, acc[1][nt]);
      acc[2][nt] = wmma_bf16(a2, b, acc[2][nt]);
      acc[3][nt] = wmma_bf16(a3, b, acc[3][nt]);
    }
  }

#pragma unroll
  for (int mt = 0; mt < 4; ++mt)
#pragma unroll
    for (int nt = 0; nt < 4; ++nt) {
      int col = col0 + nt * 16 + ln;
      float bv = bias ? bias[col] : 0.0f;
#pragma unroll
      for (int r = 0; r < 8; ++r) {
        int row = row0 + mt * 16 + r + half * 8;
        float v = acc[mt][nt][r] + bv;
        if (outb) outb[(size_t)row * ldo + col] = (bf16_t)v;
        else      outf[(size_t)row * ldo + col] = v;
      }
    }
}

// ---------------- flash attention (causal) ----------------
// One wave handles one 32-row Q tile (two 16-row WMMA row-tiles sharing the
// j-loop and the K fragments) of one (b,h).  Streams K/V in 32-column
// j-blocks with online softmax; all matmuls via v_wmma_f32_16x16x32_bf16.
__global__ void __launch_bounds__(256)
k_attn(const bf16_t* __restrict__ qkvb, const bf16_t* __restrict__ vt,
       bf16_t* __restrict__ zb) {
  __shared__ __align__(16) bf16_t pstage[8][16 * 32];  // 1KB per wave

  const int lane = threadIdx.x & 31;
  const int wave = threadIdx.x >> 5;
  const int half = lane >> 4;
  const int ln   = lane & 15;

  int tile = blockIdx.x * 8 + wave;     // (b*H+h)*64 + qt
  int qt   = tile & 63;
  int bh   = tile >> 6;
  int h    = bh & (N_HEADS - 1);
  int b    = bh >> 4;
  int q0   = qt * 32;                   // 32 query rows per wave

  const bf16_t* qptr = qkvb + (size_t)b * SEQ * QKV_N + h * HEAD_DIM;
  const bf16_t* kptr = qptr + D_MODEL;
  const bf16_t* vptr = vt + (size_t)bh * HEAD_DIM * SEQ;

  // Q A-fragments: [row-tile][d-step], kept in registers
  v16bf qa[2][2];
#pragma unroll
  for (int rt = 0; rt < 2; ++rt) {
    const bf16_t* qrow = qptr + (size_t)(q0 + rt * 16 + ln) * QKV_N;
    qa[rt][0] = load_a_frag(qrow, 0, half);
    qa[rt][1] = load_a_frag(qrow, 32, half);
  }

  v8f o[2][4];
  float m_i[2][8], l_i[2][8];
#pragma unroll
  for (int rt = 0; rt < 2; ++rt) {
#pragma unroll
    for (int dt = 0; dt < 4; ++dt) o[rt][dt] = (v8f){};
#pragma unroll
    for (int r = 0; r < 8; ++r) { m_i[rt][r] = -1e30f; l_i[rt][r] = 0.0f; }
  }

  bf16_t* pst = pstage[wave];

  for (int jt = 0; jt <= q0 + 31; jt += 32) {
    // K B-fragments for this j-block: [col-tile][d-step], shared by row-tiles
    v16bf kb[2][2];
    {
      const bf16_t* krow0 = kptr + (size_t)(jt + ln) * QKV_N;
      kb[0][0] = load_b_frag(krow0, 0, half);
      kb[0][1] = load_b_frag(krow0, 32, half);
      const bf16_t* krow1 = kptr + (size_t)(jt + 16 + ln) * QKV_N;
      kb[1][0] = load_b_frag(krow1, 0, half);
      kb[1][1] = load_b_frag(krow1, 32, half);
    }

#pragma unroll
    for (int rt = 0; rt < 2; ++rt) {
      // S = Q * K^T for columns jt..jt+31
      v8f s0 = (v8f){}, s1 = (v8f){};
      s0 = wmma_bf16(qa[rt][0], kb[0][0], s0);
      s0 = wmma_bf16(qa[rt][1], kb[0][1], s0);
      s1 = wmma_bf16(qa[rt][0], kb[1][0], s1);
      s1 = wmma_bf16(qa[rt][1], kb[1][1], s1);

      float alpha[8];
#pragma unroll
      for (int r = 0; r < 8; ++r) {
        int q = q0 + rt * 16 + r + half * 8;
        float x0 = s0[r] * 0.125f;              // 1/sqrt(64)
        float x1 = s1[r] * 0.125f;
        if (jt + ln > q)      x0 = -1e30f;      // causal mask
        if (jt + 16 + ln > q) x1 = -1e30f;
        // row max across the 16-lane half-group
        float tmax = fmaxf(x0, x1);
        for (int off = 8; off > 0; off >>= 1)
          tmax = fmaxf(tmax, __shfl_xor(tmax, off, 32));
        float mn = fmaxf(m_i[rt][r], tmax);
        alpha[r] = __expf(m_i[rt][r] - mn);
        m_i[rt][r] = mn;
        float p0 = __expf(x0 - mn);
        float p1 = __expf(x1 - mn);
        float rs = p0 + p1;
        for (int off = 8; off > 0; off >>= 1)
          rs += __shfl_xor(rs, off, 32);
        l_i[rt][r] = l_i[rt][r] * alpha[rt == 0 ? r : r] + rs;
        int row = r + half * 8;
        pst[row * 32 + ln]      = (bf16_t)p0;   // stage P tile row-major
        pst[row * 32 + 16 + ln] = (bf16_t)p1;
      }

      // rescale running O by alpha
#pragma unroll
      for (int dt = 0; dt < 4; ++dt)
#pragma unroll
        for (int r = 0; r < 8; ++r) o[rt][dt][r] *= alpha[r];

      // reload P as A-fragment (layout shuffle through wave-private LDS)
      Frag16 pf;
      pf.h[0] = *(const v8bf*)(pst + ln * 32 + half * 16);
      pf.h[1] = *(const v8bf*)(pst + ln * 32 + half * 16 + 8);

      // O += P * V  (V pre-transposed -> contiguous B-fragments)
#pragma unroll
      for (int dt = 0; dt < 4; ++dt) {
        const bf16_t* vrow = vptr + (size_t)(dt * 16 + ln) * SEQ + jt;
        Frag16 vf;
        vf.h[0] = *(const v8bf*)(vrow + half * 16);
        vf.h[1] = *(const v8bf*)(vrow + half * 16 + 8);
        o[rt][dt] = wmma_bf16(pf.v, vf.v, o[rt][dt]);
      }
    }
  }

  // normalize and store z (bf16, [NTOK, D_MODEL])
#pragma unroll
  for (int rt = 0; rt < 2; ++rt)
#pragma unroll
    for (int dt = 0; dt < 4; ++dt)
#pragma unroll
      for (int r = 0; r < 8; ++r) {
        int row = q0 + rt * 16 + r + half * 8;
        float v = o[rt][dt][r] / l_i[rt][r];
        zb[(size_t)(b * SEQ + row) * D_MODEL + h * HEAD_DIM + dt * 16 + ln] =
            (bf16_t)v;
      }
}

// ---------------- launch ----------------
extern "C" void kernel_launch(void* const* d_in, const int* in_sizes, int n_in,
                              void* d_out, int out_size, void* d_ws,
                              size_t ws_size, hipStream_t stream) {
  (void)in_sizes; (void)n_in; (void)out_size; (void)ws_size;
  const float* x     = (const float*)d_in[0];
  const float* w_qkv = (const float*)d_in[1];
  const float* b_qkv = (const float*)d_in[2];
  const float* w_out = (const float*)d_in[3];
  const float* b_out = (const float*)d_in[4];
  float* out = (float*)d_out;

  char* ws = (char*)d_ws;
  size_t off = 0;
  auto alloc = [&](size_t bytes) {
    char* p = ws + off;
    off += (bytes + 255) & ~(size_t)255;
    return p;
  };
  bf16_t* xb    = (bf16_t*)alloc((size_t)NTOK * D_MODEL * 2);
  bf16_t* wqkvT = (bf16_t*)alloc((size_t)QKV_N * D_MODEL * 2);
  bf16_t* woutT = (bf16_t*)alloc((size_t)D_MODEL * D_MODEL * 2);
  bf16_t* qkvb  = (bf16_t*)alloc((size_t)NTOK * QKV_N * 2);
  bf16_t* vt    = (bf16_t*)alloc((size_t)BATCH * N_HEADS * HEAD_DIM * SEQ * 2);
  bf16_t* zb    = (bf16_t*)alloc((size_t)NTOK * D_MODEL * 2);

  k_f32_to_bf16<<<(NTOK * D_MODEL) / 256, 256, 0, stream>>>(x, xb,
                                                            NTOK * D_MODEL);
  k_transpose_to_bf16<<<(D_MODEL * QKV_N) / 256, 256, 0, stream>>>(
      w_qkv, wqkvT, D_MODEL, QKV_N);
  k_transpose_to_bf16<<<(D_MODEL * D_MODEL) / 256, 256, 0, stream>>>(
      w_out, woutT, D_MODEL, D_MODEL);

  // QKV projection: [4096,1024] x [1024,3072] + b_qkv -> bf16 qkv
  dim3 g1(QKV_N / 64, NTOK / 512);
  k_gemm_bf16<<<g1, 256, 0, stream>>>(xb, D_MODEL, wqkvT, D_MODEL, b_qkv,
                                      qkvb, nullptr, QKV_N, D_MODEL);

  k_build_vt<<<(BATCH * N_HEADS * HEAD_DIM * SEQ) / 256, 256, 0, stream>>>(
      qkvb, vt);

  // causal flash attention: 2*16*64 wave-tiles / 8 waves per block
  k_attn<<<(BATCH * N_HEADS * (SEQ / 32)) / 8, 256, 0, stream>>>(qkvb, vt, zb);

  // output projection: [4096,1024] x [1024,1024] + b_out -> f32 out
  dim3 g2(D_MODEL / 64, NTOK / 512);
  k_gemm_bf16<<<g2, 256, 0, stream>>>(zb, D_MODEL, woutT, D_MODEL, b_out,
                                      nullptr, out, D_MODEL, D_MODEL);
}
